// MessageBlock_escn_15367392985807
// MI455X (gfx1250) — matrix operand order repaired and checked by reading, not compiled, not run on hardware
//
#include <hip/hip_runtime.h>
#include <hip/hip_bf16.h>

typedef __attribute__((ext_vector_type(16))) __bf16 v16bf;
typedef __attribute__((ext_vector_type(8)))  __bf16 v8bf;
typedef __attribute__((ext_vector_type(8)))  float  v8f;

#define NC_    9
#define C_     128
#define CIN_   256
#define EC_    16
#define NATOM  512
#define TOPK_  8
#define E_     4096
#define RESPTS 324   // 18*18 grid points
#define KPAD   352   // 324 padded up to 11*32 for K=32 WMMA steps

__constant__ int c_PERM[9] = {0,2,6,3,7,1,5,8,4};
__constant__ int c_INVP[9] = {0,5,1,3,8,6,2,4,7};
__constant__ int c_DEG[9]  = {0,1,1,1,2,2,2,2,2};

__device__ __forceinline__ float silu_f(float x){ return x / (1.0f + __expf(-x)); }
__device__ __forceinline__ __bf16 tobf(float x){ return (__bf16)x; }

// ---------------------------------------------------------------------------
// Wave-level 16x16 GEMM tile: C += A(16xK, row-major, lda) * B(Kx16) where B
// is supplied transposed (Bt: 16 rows of K, row-major, ldb). K multiple of 32.
// bf16 inputs, fp32 accumulate via v_wmma_f32_16x16x32_bf16.
// A-fragment layout (16-bit A 16x32): lane<16 -> K runs [k0,k0+8),[k0+16,k0+24)
// shifted by +8 for lanes>=16. B-fragment: lane holds column n=lane%16,
// same K runs as A.
// ---------------------------------------------------------------------------
__device__ __forceinline__ v8f wmma_tile(const __bf16* A, int lda,
                                         const __bf16* Bt, int ldb,
                                         int K, v8f acc)
{
  const int lane = threadIdx.x & 31;
  const int half = lane >> 4;
  const int l16  = lane & 15;
  const __bf16* a_row = A  + l16 * lda + half * 8;
  const __bf16* b_row = Bt + l16 * ldb + half * 8;
  for (int k0 = 0; k0 < K; k0 += 32) {
    v8bf a0 = *(const v8bf*)(a_row + k0);
    v8bf a1 = *(const v8bf*)(a_row + k0 + 16);
    v8bf b0 = *(const v8bf*)(b_row + k0);
    v8bf b1 = *(const v8bf*)(b_row + k0 + 16);
    v16bf av = __builtin_shufflevector(a0, a1, 0,1,2,3,4,5,6,7,8,9,10,11,12,13,14,15);
    v16bf bv = __builtin_shufflevector(b0, b1, 0,1,2,3,4,5,6,7,8,9,10,11,12,13,14,15);
    acc = __builtin_amdgcn_wmma_f32_16x16x32_bf16(false, av, false, bv,
                                                  (short)0, acc, false, false);
  }
  return acc;
}

// ---------------------------------------------------------------------------
// Conversions
// ---------------------------------------------------------------------------
__global__ void k_cvt_bf16(const float* __restrict__ src, __bf16* __restrict__ dst, int n){
  int i = blockIdx.x*blockDim.x + threadIdx.x;
  if (i < n) dst[i] = tobf(src[i]);
}

// src: K x N (row-major f32)  ->  dst: N x K (row-major bf16)  (i.e. B^T)
__global__ void k_transpose_bf16(const float* __restrict__ src, __bf16* __restrict__ dst,
                                 int K, int N){
  int i = blockIdx.x*blockDim.x + threadIdx.x;
  if (i < K*N) { int n = i / K, k = i - n*K; dst[i] = tobf(src[(long)k*N + n]); }
}

// Gpad: KPAD x 32 bf16 (A of stage1, K padded 9->32, rows padded 324->352)
// Ft:   16 x KPAD bf16 (A of stage2, rows i<9 valid, cols g<324 valid)
__global__ void k_grid_mats(const float* __restrict__ to_grid, const float* __restrict__ from_grid,
                            __bf16* __restrict__ Gpad, __bf16* __restrict__ Ft){
  int i = blockIdx.x*blockDim.x + threadIdx.x;
  const int nG = KPAD*32;
  if (i < nG) {
    int g = i >> 5, k = i & 31;
    Gpad[i] = (g < RESPTS && k < NC_) ? tobf(to_grid[g*NC_ + k]) : (__bf16)0.0f;
  } else if (i < nG + 16*KPAD) {
    int j = i - nG; int r = j / KPAD, g = j - r*KPAD;
    Ft[j] = (r < NC_ && g < RESPTS) ? tobf(from_grid[g*NC_ + r]) : (__bf16)0.0f;
  }
}

// ---------------------------------------------------------------------------
// h[n,i,:] = x[n,i,:] @ w_in[DEG[i]]  (+ b_in on i==0)    grid (8,32,9) x 32
// ---------------------------------------------------------------------------
__global__ void k_h_gemm(const __bf16* __restrict__ xbf, const __bf16* __restrict__ winT,
                         const float* __restrict__ b_in, float* __restrict__ h){
  int n0 = blockIdx.x*16, m0 = blockIdx.y*16, i = blockIdx.z;
  int deg = c_DEG[i];
  v8f acc = {};
  acc = wmma_tile(xbf + (long)m0*(NC_*CIN_) + i*CIN_, NC_*CIN_,
                  winT + (long)deg*(C_*CIN_) + n0*CIN_, CIN_, CIN_, acc);
  int lane = threadIdx.x & 31, half = lane>>4, l16 = lane&15;
  float bias = (i==0) ? b_in[n0 + l16] : 0.0f;
#pragma unroll
  for (int r=0;r<8;r++){
    int row = m0 + half*8 + r, col = n0 + l16;
    h[(long)row*(NC_*C_) + i*C_ + col] = acc[r] + bias;
  }
}

// ---------------------------------------------------------------------------
// Edge MLP + SO2 gates (all K=16 dots -> fp32 VALU). One block (128 thr)/edge.
// ---------------------------------------------------------------------------
__global__ void k_edge_gates(const float* __restrict__ aw,
                             const float* __restrict__ fc_dist_w, const float* __restrict__ fc_dist_b,
                             const float* __restrict__ fc_edge_w, const float* __restrict__ fc_edge_b,
                             const float* __restrict__ emb_src, const float* __restrict__ emb_tgt,
                             const int* __restrict__ an, const int* __restrict__ esrc,
                             const float* __restrict__ d0w, const float* __restrict__ d0b,
                             const float* __restrict__ m1dw, const float* __restrict__ m1db,
                             const float* __restrict__ m2dw, const float* __restrict__ m2db,
                             float* __restrict__ g0, float* __restrict__ g1, float* __restrict__ g2){
  int e = blockIdx.x, t = threadIdx.x;
  __shared__ float sa[32], su[16], xv[16];
  if (t < 32) sa[t] = aw[e*32 + t];
  __syncthreads();
  int src = esrc[e], tgt = e >> 3;
  if (t < 16) {
    float s = fc_dist_b[t];
    for (int m=0;m<32;m++) s += sa[m]*fc_dist_w[m*EC_ + t];
    s += emb_src[an[src]*EC_ + t] + emb_tgt[an[tgt]*EC_ + t];
    su[t] = silu_f(s);
  }
  __syncthreads();
  if (t < 16) {
    float s = fc_edge_b[t];
    for (int m=0;m<16;m++) s += su[m]*fc_edge_w[m*EC_ + t];
    xv[t] = silu_f(s);
  }
  __syncthreads();
  for (int b=0;b<2;b++){
    { float s = d0b[b*C_ + t];
      for (int m=0;m<16;m++) s += xv[m]*d0w[(b*EC_ + m)*C_ + t];
      g0[((long)b*E_ + e)*C_ + t] = silu_f(s); }
    for (int o = t; o < 256; o += 128) {
      float s1 = m1db[b*256 + o];
      float s2 = m2db[b*256 + o];
      for (int m=0;m<16;m++){
        s1 += xv[m]*m1dw[(b*EC_ + m)*256 + o];
        s2 += xv[m]*m2dw[(b*EC_ + m)*256 + o];
      }
      g1[((long)b*E_ + e)*256 + o] = silu_f(s1);
      g2[((long)b*E_ + e)*256 + o] = silu_f(s2);
    }
  }
}

// ---------------------------------------------------------------------------
// Wigner rotate (K=9, fp32 VALU) + PERM + bf16 cast.  One block (128 thr)/edge.
// ems[e,t,c] = bf16( sum_j wigner[e,PERM[t],j] * h[src,j,c] )  (and tgt -> emt)
// ---------------------------------------------------------------------------
__global__ void k_wigner_fwd(const float* __restrict__ wigner, const float* __restrict__ h,
                             const int* __restrict__ esrc,
                             __bf16* __restrict__ ems, __bf16* __restrict__ emt){
  int e = blockIdx.x, c = threadIdx.x;
  __shared__ float wg[81];
  if (c < 81) wg[c] = wigner[(long)e*81 + c];
  __syncthreads();
  int src = esrc[e], tgt = e >> 3;
  float hs[9], ht[9];
#pragma unroll
  for (int j=0;j<9;j++){
    hs[j] = h[(long)src*(NC_*C_) + j*C_ + c];
    ht[j] = h[(long)tgt*(NC_*C_) + j*C_ + c];
  }
#pragma unroll
  for (int t=0;t<9;t++){
    int row = c_PERM[t];
    float as = 0.f, at = 0.f;
#pragma unroll
    for (int j=0;j<9;j++){ float w = wg[row*9 + j]; as += w*hs[j]; at += w*ht[j]; }
    ems[(long)e*(NC_*C_) + t*C_ + c] = tobf(as);
    emt[(long)e*(NC_*C_) + t*C_ + c] = tobf(at);
  }
}

// ---------------------------------------------------------------------------
// SO2 GEMM-1: H[e,:] = bf16( (A[e,:] @ W1) * gate[e,:] ).  grid (8, 256) x 32.
// A rows live inside em (lda = 9*C), base already offset to the m-slot.
// ---------------------------------------------------------------------------
__global__ void k_so2_gemm1(const __bf16* __restrict__ A, const __bf16* __restrict__ Bt, int K,
                            const float* __restrict__ gate, int gate_ld, int gate_off,
                            __bf16* __restrict__ H){
  int n0 = blockIdx.x*16, e0 = blockIdx.y*16;
  v8f acc = {};
  acc = wmma_tile(A + (long)e0*(NC_*C_), NC_*C_, Bt + (long)n0*K, K, K, acc);
  int lane = threadIdx.x&31, half = lane>>4, l16 = lane&15;
#pragma unroll
  for (int r=0;r<8;r++){
    int e = e0 + half*8 + r, col = n0 + l16;
    float gv = gate[(long)e*gate_ld + gate_off + col];
    H[(long)e*C_ + col] = tobf(acc[r]*gv);
  }
}

// SO2 GEMM-2 for m=0 (plain): em_out slots 0..2.  grid (24, 256) x 32.
__global__ void k_so2_gemm2_m0(const __bf16* __restrict__ H, const __bf16* __restrict__ Bt,
                               float* __restrict__ em_out, int accumulate){
  int n0 = blockIdx.x*16, e0 = blockIdx.y*16;
  v8f acc = {};
  acc = wmma_tile(H + (long)e0*C_, C_, Bt + (long)n0*C_, C_, C_, acc);
  int lane = threadIdx.x&31, half = lane>>4, l16 = lane&15;
#pragma unroll
  for (int r=0;r<8;r++){
    int e = e0 + half*8 + r, n = n0 + l16;
    long idx = (long)e*(NC_*C_) + (n>>7)*C_ + (n&127);
    if (accumulate) em_out[idx] += acc[r]; else em_out[idx] = acc[r];
  }
}

// SO2 GEMM-2 real/imag pair: D = Hr@W2r + sign * Hi@W2i  -> slots slot_base..
__global__ void k_so2_gemm2_pair(const __bf16* __restrict__ Hr, const __bf16* __restrict__ Hi,
                                 const __bf16* __restrict__ Btr, const __bf16* __restrict__ Bti,
                                 float sign, int slot_base,
                                 float* __restrict__ em_out, int accumulate){
  int n0 = blockIdx.x*16, e0 = blockIdx.y*16;
  v8f ar = {}, ai = {};
  ar = wmma_tile(Hr + (long)e0*C_, C_, Btr + (long)n0*C_, C_, C_, ar);
  ai = wmma_tile(Hi + (long)e0*C_, C_, Bti + (long)n0*C_, C_, C_, ai);
  int lane = threadIdx.x&31, half = lane>>4, l16 = lane&15;
#pragma unroll
  for (int r=0;r<8;r++){
    int e = e0 + half*8 + r, n = n0 + l16;
    int slot = slot_base + (n>>7), col = n & 127;
    float v = ar[r] + sign*ai[r];
    long idx = (long)e*(NC_*C_) + slot*C_ + col;
    if (accumulate) em_out[idx] += v; else em_out[idx] = v;
  }
}

// ---------------------------------------------------------------------------
// Fused grid stage, one block (256 thr = 8 waves) per edge, in place on em_out:
//   X  = em[INV_PERM]             (9x128, LDS, bf16, B^T form, K padded to 32)
//   xg = silu(Gpad @ X)           streamed 32 grid rows at a time (stage1 WMMA)
//   T  = Ft @ xg                  (stage2 WMMA, acc across 11 chunks)
//   out= wigner^T @ T             (K=9, fp32 VALU)  -> overwrite em_out[e]
// ---------------------------------------------------------------------------
__global__ void __launch_bounds__(256)
k_grid(const __bf16* __restrict__ Gpad, const __bf16* __restrict__ Ft,
       const float* __restrict__ wigner, float* __restrict__ em_out){
  int e = blockIdx.x, tid = threadIdx.x;
  int wave = tid >> 5;
  int lane = tid & 31, half = lane>>4, l16 = lane&15;
  __shared__ __bf16 BtX[128*32];    // B^T of X: row=channel c, 32 K (9 used)
  __shared__ __bf16 chunk[128*32];  // xg chunk^T: row=channel c, 32 grid rows
  __shared__ float  tg[16*128];     // stage2 result, rows i (9 used) x c
  __shared__ float  wg[81];
  float* emE = em_out + (long)e*(NC_*C_);
  for (int idx = tid; idx < 128*32; idx += 256){
    int c = idx >> 5, k = idx & 31;
    BtX[idx] = (k < NC_) ? tobf(emE[c_INVP[k]*C_ + c]) : (__bf16)0.0f;
  }
  if (tid < 81) wg[tid] = wigner[(long)e*81 + tid];
  __syncthreads();
  v8f acc2 = {};
  const int n0 = wave * 16;             // this wave's 16-channel column tile
  for (int j=0;j<11;j++){
    int g0 = j*32;
#pragma unroll
    for (int mt=0; mt<2; mt++){
      v8f s = {};
      s = wmma_tile(Gpad + (long)(g0 + mt*16)*32, 32, BtX + n0*32, 32, 32, s);
#pragma unroll
      for (int r=0;r<8;r++){
        int grow = mt*16 + half*8 + r;  // local grid row 0..31
        int c = n0 + l16;
        chunk[c*32 + grow] = tobf(silu_f(s[r]));
      }
    }
    __syncthreads();
    acc2 = wmma_tile(Ft + g0, KPAD, chunk + n0*32, 32, 32, acc2);
    __syncthreads();
  }
#pragma unroll
  for (int r=0;r<8;r++) tg[(half*8 + r)*128 + n0 + l16] = acc2[r];
  __syncthreads();
  int c = tid & 127;
  float tv[9];
#pragma unroll
  for (int j=0;j<9;j++) tv[j] = tg[j*128 + c];
  for (int i = tid >> 7; i < 9; i += 2){
    float s = 0.f;
#pragma unroll
    for (int j=0;j<9;j++) s += wg[j*9 + i]*tv[j];
    emE[i*C_ + c] = s;
  }
}

// ---------------------------------------------------------------------------
// Masked top-k reduction per atom -> bf16 A for the output GEMM.
// ---------------------------------------------------------------------------
__global__ void k_sum_atoms(const float* __restrict__ xtf, const unsigned char* __restrict__ mask,
                            __bf16* __restrict__ sbf){
  int n = blockIdx.x, c = threadIdx.x;    // 128 threads
  float m[TOPK_];
#pragma unroll
  for (int k=0;k<TOPK_;k++) m[k] = mask[n*TOPK_ + k] ? 0.0f : 1.0f;
#pragma unroll
  for (int i=0;i<9;i++){
    float s = 0.f;
#pragma unroll
    for (int k=0;k<TOPK_;k++)
      s += m[k]*xtf[((long)(n*TOPK_+k))*(NC_*C_) + i*C_ + c];
    sbf[(long)n*(NC_*C_) + i*C_ + c] = tobf(s);
  }
}

// out[n,i,:] = s[n,i,:] @ w_out[DEG[i]]  (+ b_out on i==0)   grid (16,32,9) x 32
__global__ void k_out_gemm(const __bf16* __restrict__ sbf, const __bf16* __restrict__ woutT,
                           const float* __restrict__ b_out, float* __restrict__ out){
  int n0 = blockIdx.x*16, m0 = blockIdx.y*16, i = blockIdx.z;
  int deg = c_DEG[i];
  v8f acc = {};
  acc = wmma_tile(sbf + (long)m0*(NC_*C_) + i*C_, NC_*C_,
                  woutT + (long)deg*(CIN_*C_) + n0*C_, C_, C_, acc);
  int lane = threadIdx.x&31, half = lane>>4, l16 = lane&15;
  float bias = (i==0) ? b_out[n0 + l16] : 0.0f;
#pragma unroll
  for (int r=0;r<8;r++){
    int row = m0 + half*8 + r, col = n0 + l16;
    out[(long)row*(NC_*CIN_) + i*CIN_ + col] = acc[r] + bias;
  }
}

// ---------------------------------------------------------------------------
extern "C" void kernel_launch(void* const* d_in, const int* in_sizes, int n_in,
                              void* d_out, int out_size, void* d_ws, size_t ws_size,
                              hipStream_t stream) {
  const float* x         = (const float*)d_in[0];
  const float* attn_w    = (const float*)d_in[1];
  const float* wigner    = (const float*)d_in[2];
  const float* to_grid   = (const float*)d_in[3];
  const float* from_grid = (const float*)d_in[4];
  const float* w_in      = (const float*)d_in[5];
  const float* b_in      = (const float*)d_in[6];
  const float* w_out     = (const float*)d_in[7];
  const float* b_out     = (const float*)d_in[8];
  const float* emb_src   = (const float*)d_in[9];
  const float* emb_tgt   = (const float*)d_in[10];
  const float* fc_dist_w = (const float*)d_in[11];
  const float* fc_dist_b = (const float*)d_in[12];
  const float* fc_edge_w = (const float*)d_in[13];
  const float* fc_edge_b = (const float*)d_in[14];
  const float* d0w       = (const float*)d_in[15];
  const float* d0b       = (const float*)d_in[16];
  const float* m0w1      = (const float*)d_in[17];
  const float* m0w2      = (const float*)d_in[18];
  const float* m1dw      = (const float*)d_in[19];
  const float* m1db      = (const float*)d_in[20];
  const float* m1w1r     = (const float*)d_in[21];
  const float* m1w2r     = (const float*)d_in[22];
  const float* m1w1i     = (const float*)d_in[23];
  const float* m1w2i     = (const float*)d_in[24];
  const float* m2dw      = (const float*)d_in[25];
  const float* m2db      = (const float*)d_in[26];
  const float* m2w1r     = (const float*)d_in[27];
  const float* m2w2r     = (const float*)d_in[28];
  const float* m2w1i     = (const float*)d_in[29];
  const float* m2w2i     = (const float*)d_in[30];
  const int*   an        = (const int*)d_in[31];
  const int*   esrc      = (const int*)d_in[32];
  const unsigned char* amask = (const unsigned char*)d_in[33];
  float* out = (float*)d_out;

  char* ws = (char*)d_ws;
  size_t off = 0;
  auto alloc = [&](size_t bytes)->char*{
    char* p = ws + off; off = (off + bytes + 255) & ~(size_t)255; return p;
  };

  __bf16* x_bf  = (__bf16*)alloc((size_t)NATOM*NC_*CIN_*2);
  __bf16* winT  = (__bf16*)alloc((size_t)3*C_*CIN_*2);
  __bf16* woutT = (__bf16*)alloc((size_t)3*CIN_*C_*2);
  __bf16 *m0w1T[2], *m0w2T[2], *m1w1rT[2], *m1w1iT[2], *m1w2rT[2], *m1w2iT[2],
         *m2w1rT[2], *m2w1iT[2], *m2w2rT[2], *m2w2iT[2];
  for (int b=0;b<2;b++){
    m0w1T[b]=(__bf16*)alloc(128*384*2);  m0w2T[b]=(__bf16*)alloc(384*128*2);
    m1w1rT[b]=(__bf16*)alloc(128*256*2); m1w1iT[b]=(__bf16*)alloc(128*256*2);
    m1w2rT[b]=(__bf16*)alloc(256*128*2); m1w2iT[b]=(__bf16*)alloc(256*128*2);
    m2w1rT[b]=(__bf16*)alloc(128*128*2); m2w1iT[b]=(__bf16*)alloc(128*128*2);
    m2w2rT[b]=(__bf16*)alloc(128*128*2); m2w2iT[b]=(__bf16*)alloc(128*128*2);
  }
  float*  h      = (float*)alloc((size_t)NATOM*NC_*C_*4);
  float*  g0     = (float*)alloc((size_t)2*E_*C_*4);
  float*  g1     = (float*)alloc((size_t)2*E_*256*4);
  float*  g2     = (float*)alloc((size_t)2*E_*256*4);
  __bf16* ems    = (__bf16*)alloc((size_t)E_*NC_*C_*2);
  __bf16* emt    = (__bf16*)alloc((size_t)E_*NC_*C_*2);
  __bf16* hid0   = (__bf16*)alloc((size_t)E_*C_*2);
  __bf16* hid1r0 = (__bf16*)alloc((size_t)E_*C_*2);
  __bf16* hid1r1 = (__bf16*)alloc((size_t)E_*C_*2);
  __bf16* hid1i0 = (__bf16*)alloc((size_t)E_*C_*2);
  __bf16* hid1i1 = (__bf16*)alloc((size_t)E_*C_*2);
  __bf16* hid2r0 = (__bf16*)alloc((size_t)E_*C_*2);
  __bf16* hid2r1 = (__bf16*)alloc((size_t)E_*C_*2);
  __bf16* hid2i0 = (__bf16*)alloc((size_t)E_*C_*2);
  __bf16* hid2i1 = (__bf16*)alloc((size_t)E_*C_*2);
  float*  em_out = (float*)alloc((size_t)E_*NC_*C_*4);   // also final xt buffer
  __bf16* Gp     = (__bf16*)alloc((size_t)KPAD*32*2);
  __bf16* FtB    = (__bf16*)alloc((size_t)16*KPAD*2);
  __bf16* sbf    = (__bf16*)alloc((size_t)NATOM*NC_*C_*2);

  // --- weight / activation precision conversion (bf16, B^T layouts) ---------
  { int n = NATOM*NC_*CIN_;
    k_cvt_bf16<<<(n+255)/256,256,0,stream>>>(x, x_bf, n); }
  auto T = [&](const float* s, __bf16* d, int K, int N){
    k_transpose_bf16<<<(K*N+255)/256,256,0,stream>>>(s, d, K, N);
  };
  for (int dg=0;dg<3;dg++){
    T(w_in  + (size_t)dg*CIN_*C_, winT  + (size_t)dg*C_*CIN_, CIN_, C_);
    T(w_out + (size_t)dg*C_*CIN_, woutT + (size_t)dg*CIN_*C_, C_, CIN_);
  }
  for (int b=0;b<2;b++){
    T(m0w1  + (size_t)b*384*128, m0w1T[b], 384, 128);
    T(m0w2  + (size_t)b*128*384, m0w2T[b], 128, 384);
    T(m1w1r + (size_t)b*256*128, m1w1rT[b], 256, 128);
    T(m1w1i + (size_t)b*256*128, m1w1iT[b], 256, 128);
    T(m1w2r + (size_t)b*128*256, m1w2rT[b], 128, 256);
    T(m1w2i + (size_t)b*128*256, m1w2iT[b], 128, 256);
    T(m2w1r + (size_t)b*128*128, m2w1rT[b], 128, 128);
    T(m2w1i + (size_t)b*128*128, m2w1iT[b], 128, 128);
    T(m2w2r + (size_t)b*128*128, m2w2rT[b], 128, 128);
    T(m2w2i + (size_t)b*128*128, m2w2iT[b], 128, 128);
  }
  { int n = KPAD*32 + 16*KPAD;
    k_grid_mats<<<(n+255)/256,256,0,stream>>>(to_grid, from_grid, Gp, FtB); }

  // --- node embedding GEMM, edge MLP/gates, wigner rotation -----------------
  k_h_gemm<<<dim3(8,32,9),32,0,stream>>>(x_bf, winT, b_in, h);
  k_edge_gates<<<E_,128,0,stream>>>(attn_w, fc_dist_w, fc_dist_b, fc_edge_w, fc_edge_b,
                                    emb_src, emb_tgt, an, esrc,
                                    d0w, d0b, m1dw, m1db, m2dw, m2db, g0, g1, g2);
  k_wigner_fwd<<<E_,128,0,stream>>>(wigner, h, esrc, ems, emt);

  // --- SO2 blocks (b=0 on src rotation, b=1 on tgt rotation; summed) --------
  for (int b=0;b<2;b++){
    const __bf16* em = b ? emt : ems;
    const float* G0 = g0 + (size_t)b*E_*C_;
    const float* G1 = g1 + (size_t)b*E_*256;
    const float* G2 = g2 + (size_t)b*E_*256;
    dim3 g8(8,256), g16(16,256), g24(24,256);
    k_so2_gemm1<<<g8,32,0,stream>>>(em + 0*C_, m0w1T[b], 384, G0, 128, 0,   hid0);
    k_so2_gemm1<<<g8,32,0,stream>>>(em + 3*C_, m1w1rT[b], 256, G1, 256, 0,   hid1r0);
    k_so2_gemm1<<<g8,32,0,stream>>>(em + 5*C_, m1w1rT[b], 256, G1, 256, 0,   hid1r1);
    k_so2_gemm1<<<g8,32,0,stream>>>(em + 3*C_, m1w1iT[b], 256, G1, 256, 128, hid1i0);
    k_so2_gemm1<<<g8,32,0,stream>>>(em + 5*C_, m1w1iT[b], 256, G1, 256, 128, hid1i1);
    k_so2_gemm1<<<g8,32,0,stream>>>(em + 7*C_, m2w1rT[b], 128, G2, 256, 0,   hid2r0);
    k_so2_gemm1<<<g8,32,0,stream>>>(em + 8*C_, m2w1rT[b], 128, G2, 256, 0,   hid2r1);
    k_so2_gemm1<<<g8,32,0,stream>>>(em + 7*C_, m2w1iT[b], 128, G2, 256, 128, hid2i0);
    k_so2_gemm1<<<g8,32,0,stream>>>(em + 8*C_, m2w1iT[b], 128, G2, 256, 128, hid2i1);
    k_so2_gemm2_m0  <<<g24,32,0,stream>>>(hid0, m0w2T[b], em_out, b);
    k_so2_gemm2_pair<<<g16,32,0,stream>>>(hid1r0, hid1i1, m1w2rT[b], m1w2iT[b], -1.0f, 3, em_out, b);
    k_so2_gemm2_pair<<<g16,32,0,stream>>>(hid1r1, hid1i0, m1w2rT[b], m1w2iT[b], +1.0f, 5, em_out, b);
    k_so2_gemm2_pair<<<g8, 32,0,stream>>>(hid2r0, hid2i1, m2w2rT[b], m2w2iT[b], -1.0f, 7, em_out, b);
    k_so2_gemm2_pair<<<g8, 32,0,stream>>>(hid2r1, hid2i0, m2w2rT[b], m2w2iT[b], +1.0f, 8, em_out, b);
  }

  // --- fused grid activation + inverse wigner (in place on em_out) ----------
  k_grid<<<E_,256,0,stream>>>(Gp, FtB, wigner, em_out);

  // --- masked top-k reduce + output GEMM ------------------------------------
  k_sum_atoms<<<NATOM,128,0,stream>>>(em_out, amask, sbf);
  k_out_gemm<<<dim3(16,32,9),32,0,stream>>>(sbf, woutT, b_out, out);
}
